// QueryRNN_41867341201799
// MI455X (gfx1250) — matrix operand order repaired and checked by reading, not compiled, not run on hardware
//
#include <hip/hip_runtime.h>
#include <hip/hip_bf16.h>
#include <cmath>

typedef __attribute__((ext_vector_type(16))) _Float16 v16h;
typedef __attribute__((ext_vector_type(8)))  _Float16 v8h;
typedef __attribute__((ext_vector_type(8)))  float    v8f;

#define N_LEAVES 65536
#define MEM 128

__device__ __forceinline__ v8f wmma_f16(v16h a, v16h b, v8f c) {
    // v_wmma_f32_16x16x32_f16: D = A(16x32 f16) x B(32x16 f16) + C(16x16 f32)
    return __builtin_amdgcn_wmma_f32_16x16x32_f16(
        /*neg_a=*/false, a, /*neg_b=*/false, b,
        /*c_mod=*/(short)0, c, /*reuse_a=*/false, /*reuse_b=*/false);
}

// CDNA5 hardware tanh (V_TANH_F32, TRANS32). Branch-free fallbacks.
__device__ __forceinline__ float fast_tanh(float x) {
#if __has_builtin(__builtin_amdgcn_tanhf)
    return __builtin_amdgcn_tanhf(x);
#elif __has_builtin(__builtin_amdgcn_exp2f) && __has_builtin(__builtin_amdgcn_rcpf)
    float e = __builtin_amdgcn_exp2f(x * 2.88539008177792681f); // exp(2x)
    return 1.0f - 2.0f * __builtin_amdgcn_rcpf(e + 1.0f);
#else
    return tanhf(x);   // host-pass parse only
#endif
}

// ---------------------------------------------------------------------------
// f32 -> f16 conversion (grid-stride) -- weights only.
// ---------------------------------------------------------------------------
__global__ void cvt_f32_f16(const float* __restrict__ src,
                            _Float16* __restrict__ dst, int n) {
    int i = blockIdx.x * blockDim.x + threadIdx.x;
    int stride = gridDim.x * blockDim.x;
    for (; i < n; i += stride) dst[i] = (_Float16)src[i];
}

// ---------------------------------------------------------------------------
// One GEMM+tanh stage of the fused leaf MLP, per-wave 16-row strip.
//   Ydst[r, m] = tanh( sum_k Xsrc[r, k] * W[m, k] ),  r in wave's strip.
// Xsrc/Ydst live in LDS (inter-stage transpose buffer) or global memory;
// address-space inference turns LDS accesses into ds_* ops.
// Fragment layouts used here (wave32; A and B share one contiguous K order):
//   A 16x32 f16: lane<16 row=lane, halves 0..15 = K 0..15;
//                lanes>=16 shifted by K+16.
//   B 32x16 f16: lane holds column lane&15, 16 contiguous K, lanes>=16 K+16.
//   D 16x16 f32: lane holds column lane&15, VGPR j = row j + (lane<16?0:8).
// ---------------------------------------------------------------------------
template<int K, int M, bool SRC_F32>
__device__ __forceinline__ void stage_gemm(
    const void* __restrict__ Xsrc,        // row-major, leading dim K
    const _Float16* __restrict__ W,       // global, M x K row-major
    _Float16* __restrict__ Ydst,          // row-major, leading dim M
    int lane, int wave)
{
    constexpr int NK = K / 32;
    constexpr int NT = M / 16;
    const int mrow = lane & 15;
    const int hsel = (lane < 16) ? 0 : 1;
    const int rowOff = wave * 16 + mrow;

    v16h a[NK];
    if constexpr (SRC_F32) {
        const float* xrow = (const float*)Xsrc + (size_t)rowOff * K + hsel * 16;
        #pragma unroll
        for (int kt = 0; kt < NK; ++kt) {
            v8f lo = *(const v8f*)(xrow + kt * 32);
            v8f hi = *(const v8f*)(xrow + kt * 32 + 8);
            v16h t;
            #pragma unroll
            for (int e = 0; e < 8; ++e) {
                t[e]     = (_Float16)lo[e];
                t[e + 8] = (_Float16)hi[e];
            }
            a[kt] = t;
        }
    } else {
        const _Float16* xrow = (const _Float16*)Xsrc + (size_t)rowOff * K + hsel * 16;
        #pragma unroll
        for (int kt = 0; kt < NK; ++kt) {
            v8h lo = *(const v8h*)(xrow + kt * 32);        // K = kt*32 + {0|16} + 0..7
            v8h hi = *(const v8h*)(xrow + kt * 32 + 8);    // K = kt*32 + {0|16} + 8..15
            a[kt] = __builtin_shufflevector(lo, hi,
                        0,1,2,3,4,5,6,7,8,9,10,11,12,13,14,15);
        }
    }

    const int col = lane & 15;
    const int r0  = wave * 16 + hsel * 8;
    #pragma unroll
    for (int nt = 0; nt < NT; ++nt) {
        const _Float16* wrow = W + (size_t)(nt * 16 + col) * K + hsel * 16;
        v8f acc = {};
        #pragma unroll
        for (int kt = 0; kt < NK; ++kt) {
            v16h b = *(const v16h*)(wrow + kt * 32);
            acc = wmma_f16(a[kt], b, acc);
        }
        // Batched epilogue: 8 independent TRANS tanhs issue back-to-back,
        // then packed f16 converts, then b16/b16_d16_hi stores -- avoids
        // per-element TRANS32 stall chains.
        float tf[8];
        #pragma unroll
        for (int j = 0; j < 8; ++j) tf[j] = fast_tanh(acc[j]);
        v8h p;
        #pragma unroll
        for (int j = 0; j < 8; ++j) p[j] = (_Float16)tf[j];
        #pragma unroll
        for (int j = 0; j < 8; ++j) {
            Ydst[(size_t)(r0 + j) * M + (nt * 16 + col)] = p[j];
        }
    }
}

// ---------------------------------------------------------------------------
// Fused 4-stage leaf MLP: 64 -> 128 -> 384 -> 256 -> 128, tanh each stage.
// 64 rows per block (4 wave32); intermediates live entirely in LDS, so the
// only HBM traffic is the f32 input read and the (coalesced) f16 output
// write. LDS: regA (64x256, stage1+stage3), regB (64x384, stage2),
// regOut (64x128, stage4 staging) = 96KB of the WGP's 320KB.
// ---------------------------------------------------------------------------
__global__ __launch_bounds__(128) void leaf_mlp(
    const float* __restrict__ X0,        // N_LEAVES x 64 f32
    const _Float16* __restrict__ W1h,    // 128 x 64
    const _Float16* __restrict__ W2h,    // 384 x 128
    const _Float16* __restrict__ W3h,    // 256 x 384
    const _Float16* __restrict__ Wph,    // 128 x 256
    _Float16* __restrict__ Y)            // N_LEAVES x 128 f16
{
    __shared__ _Float16 regA[64 * 256];
    __shared__ _Float16 regB[64 * 384];
    __shared__ _Float16 regOut[64 * 128];
    const int lane = threadIdx.x & 31;
    const int wave = threadIdx.x >> 5;
    const size_t blockRow = (size_t)blockIdx.x * 64;

    stage_gemm< 64, 128, true >(X0 + blockRow * 64, W1h, regA, lane, wave);
    __syncthreads();
    stage_gemm<128, 384, false>(regA, W2h, regB, lane, wave);
    __syncthreads();
    stage_gemm<384, 256, false>(regB, W3h, regA, lane, wave);
    __syncthreads();
    stage_gemm<256, 128, false>(regA, Wph, regOut, lane, wave);
    __syncthreads();

    // Coalesced epilogue: 64x128 halves = 8192; 128 threads x 8 x v8h.
    // Consecutive threads cover contiguous 2KB spans -> global_store_b128.
    _Float16* dst = Y + blockRow * MEM;
    #pragma unroll
    for (int i = 0; i < 8; ++i) {
        int off = i * 1024 + threadIdx.x * 8;
        *(v8h*)(dst + off) = *(const v8h*)(regOut + off);
    }
}

// ---------------------------------------------------------------------------
// Fused tree level: for each parent p (4 children):
//   ha = mean_c tanh(child_c @ Wand^T), ho = mean_c tanh(child_c @ Wor^T)
//   Hout[p] = (ops[p]==0) ? ha : ho
// Each lane's 8 D-fragment rows are exactly 2 complete child groups, so the
// mean over children is a pure in-register reduction. Output (32 parents x
// 128) is staged in LDS and written back with coalesced b128 stores.
// All threads reach the barrier (compute is guarded, not early-returned).
// ---------------------------------------------------------------------------
__global__ __launch_bounds__(256) void tree_level(
    const _Float16* __restrict__ Hin,
    const _Float16* __restrict__ Wa, const _Float16* __restrict__ Wo,
    const int* __restrict__ ops, int opsStart,
    _Float16* __restrict__ Hout, int nOut, int NR)
{
    constexpr int K  = MEM;      // 128
    constexpr int NK = K / 32;   // 4
    constexpr int NT = MEM / 16; // 8
    __shared__ _Float16 outT[32 * MEM];   // 8KB: this block's parents
    const int lane    = threadIdx.x & 31;
    const int wave    = threadIdx.x >> 5;
    const int rowBase = (blockIdx.x * 8 + wave) * 16;
    const int pBlock  = blockIdx.x * 32;  // first parent of this block
    const bool active = rowBase < NR;

    if (active) {
        const int mrow = lane & 15;
        const int hsel = (lane < 16) ? 0 : 1;
        int arow = rowBase + mrow; if (arow >= NR) arow = NR - 1;
        const _Float16* xrow = Hin + (size_t)arow * K + hsel * 16;

        v16h a[NK];
        #pragma unroll
        for (int kt = 0; kt < NK; ++kt) {
            v8h lo = *(const v8h*)(xrow + kt * 32);
            v8h hi = *(const v8h*)(xrow + kt * 32 + 8);
            a[kt] = __builtin_shufflevector(lo, hi,
                        0,1,2,3,4,5,6,7,8,9,10,11,12,13,14,15);
        }

        const int col = lane & 15;
        const int r0  = rowBase + hsel * 8;   // first input row held by lane
        const int p0  = r0 >> 2;              // first parent (global index)
        const int pl0 = p0 - pBlock;          // parent index within block
        const int op0 = (p0     < nOut) ? ops[opsStart + p0]     : 0;
        const int op1 = (p0 + 1 < nOut) ? ops[opsStart + p0 + 1] : 0;

        #pragma unroll
        for (int nt = 0; nt < NT; ++nt) {
            const _Float16* wa = Wa + (size_t)(nt * 16 + col) * K + hsel * 16;
            const _Float16* wo = Wo + (size_t)(nt * 16 + col) * K + hsel * 16;
            v8f accA = {}, accO = {};
            #pragma unroll
            for (int kt = 0; kt < NK; ++kt) {
                v16h bA = *(const v16h*)(wa + kt * 32);
                v16h bO = *(const v16h*)(wo + kt * 32);
                accA = wmma_f16(a[kt], bA, accA);
                accO = wmma_f16(a[kt], bO, accO);
            }
            float hA0 = 0.25f * (fast_tanh(accA[0]) + fast_tanh(accA[1]) +
                                 fast_tanh(accA[2]) + fast_tanh(accA[3]));
            float hA1 = 0.25f * (fast_tanh(accA[4]) + fast_tanh(accA[5]) +
                                 fast_tanh(accA[6]) + fast_tanh(accA[7]));
            float hO0 = 0.25f * (fast_tanh(accO[0]) + fast_tanh(accO[1]) +
                                 fast_tanh(accO[2]) + fast_tanh(accO[3]));
            float hO1 = 0.25f * (fast_tanh(accO[4]) + fast_tanh(accO[5]) +
                                 fast_tanh(accO[6]) + fast_tanh(accO[7]));
            int c = nt * 16 + col;
            outT[(size_t)pl0 * MEM + c]       = (_Float16)((op0 == 0) ? hA0 : hO0);
            outT[(size_t)(pl0 + 1) * MEM + c] = (_Float16)((op1 == 0) ? hA1 : hO1);
        }
    }
    __syncthreads();

    // Coalesced write-back of this block's valid parents.
    int vp = nOut - pBlock; if (vp > 32) vp = 32;
    int totalH = vp * MEM;                    // multiple of 128
    _Float16* dst = Hout + (size_t)pBlock * MEM;
    for (int off = threadIdx.x * 8; off < totalH; off += 256 * 8) {
        *(v8h*)(dst + off) = *(const v8h*)(outT + off);
    }
}

// ---------------------------------------------------------------------------
// Readout: 1x128 vector through 3 ReLU layers + scalar head. One block.
// ---------------------------------------------------------------------------
__global__ __launch_bounds__(128) void readout(
    const _Float16* __restrict__ h,
    const float* __restrict__ Rw1, const float* __restrict__ Rb1,
    const float* __restrict__ Rw2, const float* __restrict__ Rb2,
    const float* __restrict__ Rw3, const float* __restrict__ Rb3,
    const float* __restrict__ Rwp, const float* __restrict__ Rbp,
    float* __restrict__ out)
{
    __shared__ float xs[128];
    __shared__ float ys[128];
    __shared__ float red[128];
    const int t = threadIdx.x;
    xs[t] = (float)h[t];
    __syncthreads();

    float acc = Rb1[t];
    for (int k = 0; k < 128; ++k) acc += Rw1[t * 128 + k] * xs[k];
    ys[t] = fmaxf(acc, 0.f);
    __syncthreads();

    acc = Rb2[t];
    for (int k = 0; k < 128; ++k) acc += Rw2[t * 128 + k] * ys[k];
    xs[t] = fmaxf(acc, 0.f);
    __syncthreads();

    acc = Rb3[t];
    for (int k = 0; k < 128; ++k) acc += Rw3[t * 128 + k] * xs[k];
    ys[t] = fmaxf(acc, 0.f);
    __syncthreads();

    red[t] = Rwp[t] * ys[t];
    __syncthreads();
    for (int s = 64; s > 0; s >>= 1) {
        if (t < s) red[t] += red[t + s];
        __syncthreads();
    }
    if (t == 0) out[0] = red[0] + Rbp[0];
}

// ---------------------------------------------------------------------------
extern "C" void kernel_launch(void* const* d_in, const int* in_sizes, int n_in,
                              void* d_out, int out_size, void* d_ws, size_t ws_size,
                              hipStream_t stream) {
    (void)in_sizes; (void)n_in; (void)out_size; (void)ws_size;
    const float* inputs = (const float*)d_in[0];
    const int*   ops    = (const int*)  d_in[1];
    const float* W1  = (const float*)d_in[2];
    const float* W2  = (const float*)d_in[3];
    const float* W3  = (const float*)d_in[4];
    const float* Wp  = (const float*)d_in[5];
    const float* Wand= (const float*)d_in[6];
    const float* Wor = (const float*)d_in[7];
    const float* Rw1 = (const float*)d_in[8];
    const float* Rb1 = (const float*)d_in[9];
    const float* Rw2 = (const float*)d_in[10];
    const float* Rb2 = (const float*)d_in[11];
    const float* Rw3 = (const float*)d_in[12];
    const float* Rb3 = (const float*)d_in[13];
    const float* Rwp = (const float*)d_in[14];
    const float* Rbp = (const float*)d_in[15];

    size_t off = 0;
    char* base = (char*)d_ws;
    auto take = [&](size_t bytes) -> char* {
        char* p = base + off;
        off += (bytes + 255) & ~(size_t)255;
        return p;
    };
    _Float16* bufA  = (_Float16*)take((size_t)N_LEAVES * MEM * sizeof(_Float16));
    _Float16* bufB  = (_Float16*)take((size_t)N_LEAVES * MEM * sizeof(_Float16));
    _Float16* w1h   = (_Float16*)take(128 * 64  * sizeof(_Float16));
    _Float16* w2h   = (_Float16*)take(384 * 128 * sizeof(_Float16));
    _Float16* w3h   = (_Float16*)take(256 * 384 * sizeof(_Float16));
    _Float16* wph   = (_Float16*)take(128 * 256 * sizeof(_Float16));
    _Float16* wandh = (_Float16*)take(128 * 128 * sizeof(_Float16));
    _Float16* worh  = (_Float16*)take(128 * 128 * sizeof(_Float16));

    // Convert matmul weights to f16 (tiny; stays L2-resident).
    cvt_f32_f16<<<64,  256, 0, stream>>>(W1,   w1h,   128 * 64);
    cvt_f32_f16<<<192, 256, 0, stream>>>(W2,   w2h,   384 * 128);
    cvt_f32_f16<<<384, 256, 0, stream>>>(W3,   w3h,   256 * 384);
    cvt_f32_f16<<<128, 256, 0, stream>>>(Wp,   wph,   128 * 256);
    cvt_f32_f16<<<64,  256, 0, stream>>>(Wand, wandh, 128 * 128);
    cvt_f32_f16<<<64,  256, 0, stream>>>(Wor,  worh,  128 * 128);

    // Fused leaf MLP: 65536 rows, 1024 blocks x 4 waves x 16 rows.
    leaf_mlp<<<N_LEAVES / 64, 128, 0, stream>>>(inputs, w1h, w2h, w3h, wph, bufB);

    // Tree reduction, levels 7..0 (ping-pong bufB <-> bufA).
    _Float16* cur = bufB;
    _Float16* nxt = bufA;
    for (int l = 7; l >= 0; --l) {
        int n     = 1 << (2 * l);      // parents at this level
        int NR    = n * 4;             // child rows
        int start = (n - 1) / 3;       // ops offset: (4^l - 1)/3
        int nb    = (NR + 127) / 128;
        tree_level<<<nb, 256, 0, stream>>>(cur, wandh, worh, ops, start,
                                           nxt, n, NR);
        _Float16* t = cur; cur = nxt; nxt = t;
    }

    // Readout MLP -> scalar.
    readout<<<1, 128, 0, stream>>>(cur, Rw1, Rb1, Rw2, Rb2, Rw3, Rb3,
                                   Rwp, Rbp, (float*)d_out);
}
